// regionAgg_layer_63745904607805
// MI455X (gfx1250) — matrix operand order repaired and checked by reading, not compiled, not run on hardware
//
#include <hip/hip_runtime.h>
#include <stdint.h>

typedef __bf16 bf16_t;
typedef __attribute__((ext_vector_type(16))) __bf16 v16bf;
typedef __attribute__((ext_vector_type(8)))  __bf16 v8bf;
typedef __attribute__((ext_vector_type(4)))  __bf16 v4bf;
typedef __attribute__((ext_vector_type(8)))  float  v8f;
typedef __attribute__((ext_vector_type(4)))  float  v4f;

#define BATCH   8
#define CCH     128      // channels (rows of output)
#define HWL     50176    // contraction length
#define PP      128      // parcels (cols of output)
#define SPLITS  98       // HW splits; 98 * 512 == 50176 exactly
#define KWG     512      // k-elements per workgroup
#define KC      64       // k-elements staged in LDS per chunk
#define LSTRIDE 72       // bf16 elements per LDS row (64 + 8 pad, keeps 16B align, spreads banks)

__global__ void zero_out_kernel(float* __restrict__ out, int n) {
    int i = blockIdx.x * blockDim.x + threadIdx.x;
    if (i < n) out[i] = 0.0f;
}

__global__ __launch_bounds__(256)
void parcel_gemm_kernel(const float* __restrict__ x,
                        const float* __restrict__ mask,
                        float* __restrict__ out)
{
    __shared__ bf16_t sXhi[CCH * LSTRIDE];
    __shared__ bf16_t sXlo[CCH * LSTRIDE];
    __shared__ bf16_t sM [PP  * LSTRIDE];

    const int t     = threadIdx.x;
    const int lane  = t & 31;
    const int wave  = t >> 5;          // 8 waves; wave owns output rows [wave*16, wave*16+16)
    const int half  = lane >> 4;       // wave32 half selector for WMMA fragment layout
    const int l15   = lane & 15;
    const int b     = blockIdx.y;
    const int kbase = blockIdx.x * KWG;

    const float* xb = x    + (size_t)b * CCH * HWL;
    const float* mb = mask + (size_t)b * PP  * HWL;

    v8f acc[8] = {};   // 8 tiles of 16x16 f32 along P

    for (int chunk = 0; chunk < KWG / KC; ++chunk) {
        const int k0 = kbase + chunk * KC;

        __syncthreads();   // previous chunk's compute done before overwrite

        // ---- stage 128x64 f32 of x (split into bf16 hi/lo) and 128x64 of mask into LDS ----
        // 2048 float4 per matrix / 256 threads = 8 each; 16 consecutive lanes cover one row.
        #pragma unroll
        for (int i = 0; i < 8; ++i) {
            const int idx = i * 256 + t;        // 0..2047
            const int row = idx >> 4;           // 0..127
            const int q   = (idx & 15) << 2;    // 0,4,..,60

            v4f xv = *(const v4f*)(xb + (size_t)row * HWL + k0 + q);
            v4bf hi, lo;
            #pragma unroll
            for (int e = 0; e < 4; ++e) {
                bf16_t h = (bf16_t)xv[e];
                hi[e] = h;
                lo[e] = (bf16_t)(xv[e] - (float)h);
            }
            *(v4bf*)(sXhi + row * LSTRIDE + q) = hi;
            *(v4bf*)(sXlo + row * LSTRIDE + q) = lo;

            v4f mv = *(const v4f*)(mb + (size_t)row * HWL + k0 + q);
            v4bf mm;
            #pragma unroll
            for (int e = 0; e < 4; ++e) mm[e] = (bf16_t)mv[e];
            *(v4bf*)(sM + row * LSTRIDE + q) = mm;
        }

        __syncthreads();   // staged data visible

        // ---- compute: 2 k-steps of 32, 8 P-tiles, hi+lo wmma each ----
        #pragma unroll
        for (int kk = 0; kk < KC; kk += 32) {
            // A fragment (16x32 bf16): lane(0-15)=row M, half selects K 8h+{0..7} and 16+8h+{0..7}
            const int arow  = wave * 16 + l15;
            const int abase = arow * LSTRIDE + kk + 8 * half;
            union { v16bf v; v8bf h[2]; } ahi, alo;
            ahi.h[0] = *(const v8bf*)(sXhi + abase);
            ahi.h[1] = *(const v8bf*)(sXhi + abase + 16);
            alo.h[0] = *(const v8bf*)(sXlo + abase);
            alo.h[1] = *(const v8bf*)(sXlo + abase + 16);

            #pragma unroll
            for (int tile = 0; tile < 8; ++tile) {
                // B fragment (32x16 bf16): lane(0-15)=col N, half selects K 8h+{0..7} and 16+8h+{0..7}
                const int prow  = tile * 16 + l15;
                const int bbase = prow * LSTRIDE + kk + 8 * half;
                union { v16bf v; v8bf h[2]; } bm;
                bm.h[0] = *(const v8bf*)(sM + bbase);
                bm.h[1] = *(const v8bf*)(sM + bbase + 16);

                acc[tile] = __builtin_amdgcn_wmma_f32_16x16x32_bf16(
                    false, ahi.v, false, bm.v, (short)0, acc[tile], false, false);
                acc[tile] = __builtin_amdgcn_wmma_f32_16x16x32_bf16(
                    false, alo.v, false, bm.v, (short)0, acc[tile], false, false);
            }
        }
    }

    // ---- commit partial 128x128 tile with hardware f32 atomics (out is L2-resident) ----
    // C/D layout: VGPR v -> M = v + 8*half ; lane(0-15) -> N
    float* outb = out + (size_t)b * CCH * PP;
    #pragma unroll
    for (int tile = 0; tile < 8; ++tile) {
        const int p = tile * 16 + l15;
        #pragma unroll
        for (int v = 0; v < 8; ++v) {
            const int c = wave * 16 + v + 8 * half;
            float val = acc[tile][v];
            float* addr = outb + c * PP + p;
            asm volatile("global_atomic_add_f32 %0, %1, off"
                         :: "v"((uint64_t)(uintptr_t)addr), "v"(val)
                         : "memory");
        }
    }
}

extern "C" void kernel_launch(void* const* d_in, const int* in_sizes, int n_in,
                              void* d_out, int out_size, void* d_ws, size_t ws_size,
                              hipStream_t stream) {
    const float* x    = (const float*)d_in[0];
    const float* mask = (const float*)d_in[1];
    float* out        = (float*)d_out;

    zero_out_kernel<<<(out_size + 255) / 256, 256, 0, stream>>>(out, out_size);

    dim3 grid(SPLITS, BATCH, 1);
    parcel_gemm_kernel<<<grid, 256, 0, stream>>>(x, mask, out);
}
